// _ManualLSTMCell_5171140624835
// MI455X (gfx1250) — matrix operand order repaired and checked by reading, not compiled, not run on hardware
//
#include <hip/hip_runtime.h>

// ---- WMMA types (CDNA5 / gfx1250, wave32) ----
typedef __attribute__((ext_vector_type(16))) __bf16 v16bf;
typedef __attribute__((ext_vector_type(8)))  __bf16 v8bf;
typedef __attribute__((ext_vector_type(4)))  __bf16 v4bf;
typedef __attribute__((ext_vector_type(8)))  float  v8f;

union FragBF { v16bf v; v8bf h[2]; };

#define BATCH   4096
#define KDIM    1024   // IN_DIM == HID == 1024
#define HID_    1024
#define BM      128    // block rows (batch)
#define BN      64     // block cols per gate
#define KT      32     // K per WMMA step

__device__ __forceinline__ float fast_sigmoid(float x) {
    return 1.0f / (1.0f + __expf(-x));
}

__global__ __launch_bounds__(256)
void lstm_fused_wmma_kernel(const float* __restrict__ x_t,
                            const float* __restrict__ h_prev,
                            const float* __restrict__ c_prev,
                            const float* __restrict__ U_w,
                            const float* __restrict__ U_b,
                            const float* __restrict__ W_w,
                            float* __restrict__ out_h,
                            float* __restrict__ out_c) {
    // LDS staging tiles, bf16, double buffered.
    // A: BM x KT (row-major, row stride KT). B: 4 gates * BN rows x KT.
    __shared__ __attribute__((aligned(16))) __bf16 As[2][BM * KT];        // 2 * 8KB
    __shared__ __attribute__((aligned(16))) __bf16 Bs[2][4 * BN * KT];    // 2 * 16KB

    const int tid    = threadIdx.x;
    const int lane   = tid & 31;
    const int wave   = tid >> 5;       // 0..7
    const int wave_m = wave & 3;       // 4 waves along M (32 rows each)
    const int wave_n = wave >> 2;      // 2 waves along N (32 cols each)
    const int lo     = lane & 15;
    const int hi     = lane >> 4;      // 0 or 1

    const int blockRow = blockIdx.x;           // 0..31  (rows of 128)
    const int n0       = blockIdx.y * BN;      // 0..960 (cols per gate)

    // 4 gates x 2 m-frags x 2 n-frags fp32 accumulators
    v8f acc[4][2][2] = {};

    // Cooperative load of one K-step into LDS buffer `buf`.
    auto loadStep = [&](int kk, int buf) {
        const int phase = kk >> 5;             // 0: x*U^T, 1: h*W^T
        const int k0    = (kk & 31) * KT;
        const float* __restrict__ Ag = (phase ? h_prev : x_t);
        const float* __restrict__ Wg = (phase ? W_w : U_w);
        const size_t arow0 = (size_t)blockRow * BM;

        // A tile: 128 rows x 32 floats = 1024 float4s, 4 per thread
        #pragma unroll
        for (int i = 0; i < 4; ++i) {
            const int idx = i * 256 + tid;         // 0..1023
            const int r   = idx >> 3;              // 0..127
            const int c4  = (idx & 7) * 4;         // 0..28
            const float4 f = *(const float4*)(Ag + (arow0 + r) * KDIM + k0 + c4);
            v4bf pk = { (__bf16)f.x, (__bf16)f.y, (__bf16)f.z, (__bf16)f.w };
            *(v4bf*)&As[buf][r * KT + c4] = pk;    // ds_store_b64
        }
        // B tiles: 4 gates x 64 rows x 32 floats = 2048 float4s, 8 per thread
        #pragma unroll
        for (int i = 0; i < 8; ++i) {
            const int idx  = i * 256 + tid;        // 0..2047
            const int r    = idx >> 3;             // 0..255 (gate-major)
            const int c4   = (idx & 7) * 4;
            const int gate = r >> 6;
            const int nr   = r & 63;
            const float4 f = *(const float4*)(Wg + (size_t)(gate * HID_ + n0 + nr) * KDIM + k0 + c4);
            v4bf pk = { (__bf16)f.x, (__bf16)f.y, (__bf16)f.z, (__bf16)f.w };
            *(v4bf*)&Bs[buf][r * KT + c4] = pk;
        }
    };

    loadStep(0, 0);
    __syncthreads();

    for (int kk = 0; kk < 64; ++kk) {
        const int cur = kk & 1;
        if (kk + 1 < 64) loadStep(kk + 1, cur ^ 1);   // prefetch into other buffer

        // A fragments: 16-bit A 16x32 layout — lanes<16: K0..7,K16..23; lanes>=16: K8..15,K24..31
        FragBF af[2];
        #pragma unroll
        for (int mi = 0; mi < 2; ++mi) {
            const __bf16* p = &As[cur][(wave_m * 32 + mi * 16 + lo) * KT + hi * 8];
            af[mi].h[0] = *(const v8bf*)p;           // ds_load_b128
            af[mi].h[1] = *(const v8bf*)(p + 16);
        }

        #pragma unroll
        for (int g = 0; g < 4; ++g) {
            // B fragments: lane = column N; lanes<16 hold K0..7,K16..23; lanes>=16 hold K8..15,K24..31
            FragBF bfr[2];
            #pragma unroll
            for (int ni = 0; ni < 2; ++ni) {
                const __bf16* p = &Bs[cur][(g * BN + wave_n * 32 + ni * 16 + lo) * KT + hi * 8];
                bfr[ni].h[0] = *(const v8bf*)p;
                bfr[ni].h[1] = *(const v8bf*)(p + 16);
            }
            #pragma unroll
            for (int mi = 0; mi < 2; ++mi)
                #pragma unroll
                for (int ni = 0; ni < 2; ++ni)
                    acc[g][mi][ni] = __builtin_amdgcn_wmma_f32_16x16x32_bf16(
                        false, af[mi].v, false, bfr[ni].v,
                        (short)0, acc[g][mi][ni], false, false);
        }
        __syncthreads();
    }

    // Fused epilogue: bias + gates + cell update, write h_t / c_t.
    // C/D layout: VGPR r -> row (r + hi*8), col = lo within 16x16 tile.
    const int colbase = n0 + wave_n * 32;
    #pragma unroll
    for (int mi = 0; mi < 2; ++mi) {
        #pragma unroll
        for (int ni = 0; ni < 2; ++ni) {
            const int col = colbase + ni * 16 + lo;
            const float bF = U_b[col];
            const float bI = U_b[1 * HID_ + col];
            const float bO = U_b[2 * HID_ + col];
            const float bG = U_b[3 * HID_ + col];
            const int rowb = blockRow * BM + wave_m * 32 + mi * 16 + hi * 8;
            #pragma unroll
            for (int r = 0; r < 8; ++r) {
                const int row = rowb + r;
                const float fpre = acc[0][mi][ni][r] + bF;
                const float ipre = acc[1][mi][ni][r] + bI;
                const float opre = acc[2][mi][ni][r] + bO;
                const float gpre = acc[3][mi][ni][r] + bG;
                const float ft = fast_sigmoid(fpre);
                const float it = fast_sigmoid(ipre);
                const float ot = fast_sigmoid(opre);
                const float gt = tanhf(gpre);
                const float cp = c_prev[(size_t)row * HID_ + col];
                const float ct = ft * cp + it * gt;
                const float ht = ot * tanhf(ct);
                out_h[(size_t)row * HID_ + col] = ht;
                out_c[(size_t)row * HID_ + col] = ct;
            }
        }
    }
}

extern "C" void kernel_launch(void* const* d_in, const int* in_sizes, int n_in,
                              void* d_out, int out_size, void* d_ws, size_t ws_size,
                              hipStream_t stream) {
    (void)in_sizes; (void)n_in; (void)out_size; (void)d_ws; (void)ws_size;
    const float* x_t    = (const float*)d_in[0];
    const float* h_prev = (const float*)d_in[1];
    const float* c_prev = (const float*)d_in[2];
    const float* U_w    = (const float*)d_in[3];
    const float* U_b    = (const float*)d_in[4];
    const float* W_w    = (const float*)d_in[5];
    float* out_h = (float*)d_out;
    float* out_c = out_h + (size_t)BATCH * HID_;

    dim3 grid(BATCH / BM, HID_ / BN);   // 32 x 16 = 512 blocks
    dim3 block(256);                    // 8 wave32 waves
    lstm_fused_wmma_kernel<<<grid, block, 0, stream>>>(
        x_t, h_prev, c_prev, U_w, U_b, W_w, out_h, out_c);
}
